// SSMLayer_53815940219361
// MI455X (gfx1250) — matrix operand order repaired and checked by reading, not compiled, not run on hardware
//
#include <hip/hip_runtime.h>
#include <hip/hip_bf16.h>
#include <math.h>
#include <cstdint>

// ---------------------------------------------------------------------------
// Mamba SSM layer for gfx1250 (MI455X).
//   B=2, L=2048, D_MODEL=1024, D_INNER=2048, D_STATE=16, D_CONV=4, DT_RANK=64
// Pipeline:
//   1) xz      = x @ in_proj_w^T            (LDS-tiled WMMA bf16, 4096x4096x1024)
//   2) xs_conv = silu(causal_dwconv(xs)+b)  (elementwise)
//   3) x_dbl   = xs_conv @ x_proj_w^T       (direct WMMA, N=96, K=2048)
//   4) dt      = softplus(dt_r @ dt_proj_w^T + b) (LDS-tiled WMMA, N=2048,K=64)
//   5) selective scan fused with D-skip + silu(z) gating  (serial L, v_exp_f32)
//   6) out     = y @ out_proj_w^T           (LDS-tiled WMMA, N=1024,K=2048)
// Tiled GEMM: 64x64 block tile, BK=32, double-buffered LDS staged with
// GLOBAL_LOAD_ASYNC_TO_LDS_B128 (ASYNCcnt) when available.
// ---------------------------------------------------------------------------

#define B_SZ     2
#define L_SEQ    2048
#define D_MODEL  1024
#define D_INNER  2048
#define D_STATE  16
#define D_CONV   4
#define DT_RANK  64
#define MROWS    (B_SZ * L_SEQ)          /* 4096 */

typedef __attribute__((ext_vector_type(16))) __bf16 v16bf;
typedef __attribute__((ext_vector_type(8)))  float  v8f;
typedef int v4i32 __attribute__((vector_size(16)));   // matches builtin proto

#if __has_builtin(__builtin_amdgcn_global_load_async_to_lds_b128)
#define HAVE_ASYNC_LDS 1
#else
#define HAVE_ASYNC_LDS 0
#endif

__device__ __forceinline__ void async_wait(int pending) {
#if HAVE_ASYNC_LDS
#if __has_builtin(__builtin_amdgcn_s_wait_asynccnt)
    if (pending) __builtin_amdgcn_s_wait_asynccnt(4);
    else         __builtin_amdgcn_s_wait_asynccnt(0);
#else
    if (pending) asm volatile("s_wait_asynccnt 0x4" ::: "memory");
    else         asm volatile("s_wait_asynccnt 0x0" ::: "memory");
#endif
#else
    (void)pending;
#endif
}

// Copy one 16-byte chunk global -> LDS (async DMA path when available).
__device__ __forceinline__ void stage_copy(float* ldsDst, const float* gSrc) {
#if HAVE_ASYNC_LDS
    __builtin_amdgcn_global_load_async_to_lds_b128(
        (__attribute__((address_space(1))) v4i32*)(uintptr_t)gSrc,
        (__attribute__((address_space(3))) v4i32*)(uint32_t)(uintptr_t)ldsDst,
        0, 0);
#else
    *(float4*)ldsDst = *(const float4*)gSrc;     // ds_store_b128 fallback
#endif
}

__device__ __forceinline__ void cvt4(v16bf& v, int i, float4 f) {
    v[i + 0] = (__bf16)f.x;
    v[i + 1] = (__bf16)f.y;
    v[i + 2] = (__bf16)f.z;
    v[i + 3] = (__bf16)f.w;
}

// Build a 16-element bf16 fragment from fp32 p[0..7] and p[16..23].
__device__ __forceinline__ v16bf frag16(const float* p) {
    v16bf f;
    float4 q0 = *(const float4*)(p + 0);
    float4 q1 = *(const float4*)(p + 4);
    float4 q2 = *(const float4*)(p + 16);
    float4 q3 = *(const float4*)(p + 20);
    cvt4(f, 0, q0); cvt4(f, 4, q1); cvt4(f, 8, q2); cvt4(f, 12, q3);
    return f;
}

__device__ __forceinline__ float softplusf(float v) {
    return (v > 15.0f) ? v : log1pf(__expf(v));
}

// ---------------------------------------------------------------------------
// LDS-tiled WMMA GEMM:  C[M,N] = A[M,K(lda)] * W[N,K(ldw)]^T  (+ epilogue)
// Block: 256 threads (8 waves), BM=BN=64, BK=32, double-buffered LDS.
// Wave (wm = w&3, wn = w>>2) computes a 16x32 strip: 2 WMMAs sharing af.
// Requires M,N multiples of 64; K multiple of 32.
// Fragment layouts per CDNA5 ISA 16-bit WMMA VGPR mapping (A 16x32 / B 32x16).
// ---------------------------------------------------------------------------
template <int EPI>   // 0 = plain store, 1 = bias + softplus
__global__ __launch_bounds__(256) void wmma_gemm_tiled(
    const float* __restrict__ A, const float* __restrict__ W,
    const float* __restrict__ bias, float* __restrict__ C,
    int M, int N, int K, int lda, int ldw, int ldc)
{
    __shared__ float ldsA[2][64 * 32];
    __shared__ float ldsB[2][64 * 32];

    const int t     = threadIdx.x;
    const int lane  = t & 31;
    const int wave  = t >> 5;
    const int lan   = lane & 15;
    const int half  = lane >> 4;
    const int wm    = wave & 3;          // 0..3 -> 16-row group
    const int wn    = wave >> 2;         // 0..1 -> 32-col group

    const int tilesN = N >> 6;
    const int bm = blockIdx.x / tilesN;
    const int bn = blockIdx.x % tilesN;
    const int m0 = bm << 6;
    const int n0 = bn << 6;

    const int kbA = half * 8;            // A lane K-offset
    const int kbB = half * 8;            // B lane K-offset
    const int nk  = K >> 5;

    // stage loader: 64 rows x 32 floats for A and for W, 512 16B-chunks each;
    // each thread moves 2 chunks of A + 2 chunks of B (4 async ops / wave).
    auto load_stage = [&](int s, int k0) {
#pragma unroll
        for (int c = t; c < 512; c += 256) {
            const int row = c >> 3;
            const int kc  = (c & 7) << 2;
            stage_copy(&ldsA[s][row * 32 + kc],
                       A + (size_t)(m0 + row) * lda + k0 + kc);
            stage_copy(&ldsB[s][row * 32 + kc],
                       W + (size_t)(n0 + row) * ldw + k0 + kc);
        }
    };

    load_stage(0, 0);

    v8f acc0 = {}, acc1 = {};
    for (int i = 0; i < nk; ++i) {
        const int cur = i & 1;
        const int more = (i + 1 < nk);
        if (more) load_stage(cur ^ 1, (i + 1) << 5);
        async_wait(more);                 // my stage-`cur` copies are done
        __syncthreads();                  // everyone's copies done / reads done

        const float* Arow = &ldsA[cur][(wm * 16 + lan) * 32];
        const float* B0   = &ldsB[cur][(wn * 32 + lan) * 32 + kbB];
        const float* B1   = B0 + 16 * 32;

        v16bf af;
        {
            float4 a0 = *(const float4*)(Arow + kbA);
            float4 a1 = *(const float4*)(Arow + kbA + 4);
            float4 a2 = *(const float4*)(Arow + 16 + kbA);
            float4 a3 = *(const float4*)(Arow + 16 + kbA + 4);
            cvt4(af, 0, a0); cvt4(af, 4, a1); cvt4(af, 8, a2); cvt4(af, 12, a3);
        }
        v16bf bf0 = frag16(B0);
        v16bf bf1 = frag16(B1);

        acc0 = __builtin_amdgcn_wmma_f32_16x16x32_bf16(
            false, af, false, bf0, (short)0, acc0, false, false);
        acc1 = __builtin_amdgcn_wmma_f32_16x16x32_bf16(
            false, af, false, bf1, (short)0, acc1, false, false);

        __syncthreads();                  // stage `cur` free for reload
    }

    const int row0 = m0 + wm * 16 + half * 8;
    const int col0 = n0 + wn * 32 + lan;
    const int col1 = col0 + 16;
#pragma unroll
    for (int r = 0; r < 8; ++r) {
        float v0 = acc0[r], v1 = acc1[r];
        if (EPI == 1) {
            v0 = softplusf(v0 + bias[col0]);
            v1 = softplusf(v1 + bias[col1]);
        }
        C[(size_t)(row0 + r) * ldc + col0] = v0;
        C[(size_t)(row0 + r) * ldc + col1] = v1;
    }
}

// ---------------------------------------------------------------------------
// Direct per-wave-tile WMMA GEMM for skinny N (x_proj: N=96).
// ---------------------------------------------------------------------------
__global__ __launch_bounds__(256) void wmma_gemm_skinny(
    const float* __restrict__ A, const float* __restrict__ W,
    float* __restrict__ C, int M, int N, int K, int lda, int ldw, int ldc)
{
    const int lane  = threadIdx.x & 31;
    const int wave  = threadIdx.x >> 5;
    const int tilesN = N >> 4;
    const int totalTiles = (M >> 4) * tilesN;
    const int tile = blockIdx.x * 8 + wave;      // wave-uniform
    if (tile >= totalTiles) return;

    const int tm = tile / tilesN, tn = tile % tilesN;
    const int half = lane >> 4, lan = lane & 15;
    const float* Arow = A + (size_t)(tm * 16 + lan) * lda;
    const float* Wrow = W + (size_t)(tn * 16 + lan) * ldw;
    const int kbA = half * 8, kbB = half * 8;

    v8f acc = {};
    for (int k0 = 0; k0 < K; k0 += 32) {
        if (k0 + 32 < K) {
            __builtin_prefetch(Arow + k0 + 32 + kbA, 0, 3);
            __builtin_prefetch(Wrow + k0 + 32 + kbB, 0, 3);
        }
        v16bf af;
        {
            float4 a0 = *(const float4*)(Arow + k0 + kbA);
            float4 a1 = *(const float4*)(Arow + k0 + kbA + 4);
            float4 a2 = *(const float4*)(Arow + k0 + 16 + kbA);
            float4 a3 = *(const float4*)(Arow + k0 + 16 + kbA + 4);
            cvt4(af, 0, a0); cvt4(af, 4, a1); cvt4(af, 8, a2); cvt4(af, 12, a3);
        }
        v16bf bf = frag16(Wrow + k0 + kbB);
        acc = __builtin_amdgcn_wmma_f32_16x16x32_bf16(
            false, af, false, bf, (short)0, acc, false, false);
    }

    const int row0 = tm * 16 + half * 8;
    const int col  = tn * 16 + lan;
#pragma unroll
    for (int r = 0; r < 8; ++r)
        C[(size_t)(row0 + r) * ldc + col] = acc[r];
}

// ---------------------------------------------------------------------------
// Causal depthwise conv (width 4) + bias + SiLU.
// ---------------------------------------------------------------------------
__global__ __launch_bounds__(256) void conv_silu_kernel(
    const float* __restrict__ xz, const float* __restrict__ conv_w,
    const float* __restrict__ conv_b, float* __restrict__ xs_conv)
{
    const int idx = blockIdx.x * 256 + threadIdx.x;
    if (idx >= MROWS * D_INNER) return;
    const int d = idx % D_INNER;
    const int m = idx / D_INNER;
    const int l = m % L_SEQ;
    const int mbase = m - l;

    float acc = conv_b[d];
#pragma unroll
    for (int j = 0; j < D_CONV; ++j) {
        const int ls = l - (D_CONV - 1) + j;
        if (ls >= 0)
            acc += conv_w[d * D_CONV + j] *
                   xz[(size_t)(mbase + ls) * (2 * D_INNER) + d];
    }
    const float sig = 1.0f / (1.0f + __expf(-acc));
    xs_conv[(size_t)m * D_INNER + d] = acc * sig;
}

// ---------------------------------------------------------------------------
// Selective scan: one thread per (b,d); 16-state h in VGPRs; fuses D-skip
// and silu(z) gating.  y aliases xs_conv (read-before-write per element).
// ---------------------------------------------------------------------------
__global__ __launch_bounds__(256) void scan_kernel(
    const float* __restrict__ xz, const float* __restrict__ xs_conv,
    const float* __restrict__ x_dbl, const float* __restrict__ dt,
    const float* __restrict__ A_log, const float* __restrict__ Dp,
    float* __restrict__ y)
{
    const int idx = blockIdx.x * 256 + threadIdx.x;
    if (idx >= B_SZ * D_INNER) return;
    const int d = idx % D_INNER;
    const int b = idx / D_INNER;

    float Ar[D_STATE], h[D_STATE];
#pragma unroll
    for (int n = 0; n < D_STATE; ++n) {
        Ar[n] = -__expf(A_log[d * D_STATE + n]);
        h[n]  = 0.0f;
    }
    const float Dd = Dp[d];

    for (int t = 0; t < L_SEQ; ++t) {
        const size_t m  = (size_t)b * L_SEQ + t;
        const float dtv = dt[m * D_INNER + d];
        const float xv  = xs_conv[m * D_INNER + d];
        const float zv  = xz[m * (2 * D_INNER) + D_INNER + d];
        const float* __restrict__ bc =
            x_dbl + m * (DT_RANK + 2 * D_STATE) + DT_RANK;
        const float du = dtv * xv;
        float yv = 0.0f;
#pragma unroll
        for (int n = 0; n < D_STATE; ++n) {
            const float dA = __expf(dtv * Ar[n]);     // v_exp_f32
            h[n] = dA * h[n] + du * bc[n];
            yv  += h[n] * bc[D_STATE + n];
        }
        yv += Dd * xv;
        const float sz = zv / (1.0f + __expf(-zv));
        y[m * D_INNER + d] = yv * sz;
    }
}

// ---------------------------------------------------------------------------
extern "C" void kernel_launch(void* const* d_in, const int* in_sizes, int n_in,
                              void* d_out, int out_size, void* d_ws, size_t ws_size,
                              hipStream_t stream)
{
    (void)in_sizes; (void)n_in; (void)out_size; (void)ws_size;
    const float* x         = (const float*)d_in[0];
    const float* in_proj_w = (const float*)d_in[1];
    const float* conv_w    = (const float*)d_in[2];
    const float* conv_b    = (const float*)d_in[3];
    const float* x_proj_w  = (const float*)d_in[4];
    const float* dt_proj_w = (const float*)d_in[5];
    const float* dt_proj_b = (const float*)d_in[6];
    const float* A_log     = (const float*)d_in[7];
    const float* Dp        = (const float*)d_in[8];
    const float* out_proj_w= (const float*)d_in[9];
    float* out = (float*)d_out;

    // Workspace layout (floats): total = 33,947,648 floats ~= 135.8 MB.
    float* ws      = (float*)d_ws;
    float* xz      = ws;                                       // 4096*4096
    float* xs_conv = xz      + (size_t)MROWS * 2 * D_INNER;    // 4096*2048
    float* x_dbl   = xs_conv + (size_t)MROWS * D_INNER;        // 4096*96
    float* dt      = x_dbl   + (size_t)MROWS * (DT_RANK + 2 * D_STATE);
    float* y       = xs_conv;   // safe alias: read-before-write per element

    const int NDBL = DT_RANK + 2 * D_STATE;                    // 96

    // 1) xz = x @ in_proj_w^T : M=4096, N=4096, K=1024
    wmma_gemm_tiled<0><<<(MROWS / 64) * ((2 * D_INNER) / 64), 256, 0, stream>>>(
        x, in_proj_w, nullptr, xz,
        MROWS, 2 * D_INNER, D_MODEL, D_MODEL, D_MODEL, 2 * D_INNER);

    // 2) causal depthwise conv + silu
    conv_silu_kernel<<<(MROWS * D_INNER + 255) / 256, 256, 0, stream>>>(
        xz, conv_w, conv_b, xs_conv);

    // 3) x_dbl = xs_conv @ x_proj_w^T : M=4096, N=96, K=2048 (skinny path)
    {
        int tiles = (MROWS / 16) * (NDBL / 16);
        wmma_gemm_skinny<<<(tiles + 7) / 8, 256, 0, stream>>>(
            xs_conv, x_proj_w, x_dbl,
            MROWS, NDBL, D_INNER, D_INNER, D_INNER, NDBL);
    }

    // 4) dt = softplus(x_dbl[:, :64] @ dt_proj_w^T + b) : M=4096, N=2048, K=64
    wmma_gemm_tiled<1><<<(MROWS / 64) * (D_INNER / 64), 256, 0, stream>>>(
        x_dbl, dt_proj_w, dt_proj_b, dt,
        MROWS, D_INNER, DT_RANK, NDBL, DT_RANK, D_INNER);

    // 5) selective scan fused with D-skip + silu(z) gating
    scan_kernel<<<(B_SZ * D_INNER + 255) / 256, 256, 0, stream>>>(
        xz, xs_conv, x_dbl, dt, A_log, Dp, y);

    // 6) out = y @ out_proj_w^T : M=4096, N=1024, K=2048
    wmma_gemm_tiled<0><<<(MROWS / 64) * (D_MODEL / 64), 256, 0, stream>>>(
        y, out_proj_w, nullptr, out,
        MROWS, D_MODEL, D_INNER, D_INNER, D_INNER, D_MODEL);
}
